// DSMVPool_46961172415329
// MI455X (gfx1250) — compile-verified, matrix operand order checked by reading, not run on hardware
//
#include <hip/hip_runtime.h>
#include <stdint.h>

#define Nn 100000
#define Ee 1600000
#define Dd 128
#define Cc 256
#define HIDn 64
#define Kk 80000
#define NBINS 65536
#define WSTR 136   // padded LDS row stride in u16 (68 dwords -> conflict-free col reads)

typedef __attribute__((ext_vector_type(16))) __bf16 v16bf;
typedef __attribute__((ext_vector_type(8)))  float  v8f;
typedef __attribute__((ext_vector_type(4)))  unsigned int u32x4;

union FragBf { v16bf bf; u32x4 u[2]; };

static __device__ __forceinline__ unsigned short f2bf(float f) {
  unsigned u = __float_as_uint(f);
  u += 0x7FFFu + ((u >> 16) & 1u);           // round-to-nearest-even
  return (unsigned short)(u >> 16);
}
static __device__ __forceinline__ unsigned encOrd(float f) {
  unsigned u = __float_as_uint(f);
  return (u & 0x80000000u) ? ~u : (u | 0x80000000u);   // monotonic float->u32
}
static __device__ __forceinline__ float decOrd(unsigned u) {
  unsigned v = (u & 0x80000000u) ? (u & 0x7FFFFFFFu) : ~u;
  return __uint_as_float(v);
}
static __device__ __forceinline__ float waveRed32(float p) {
  #pragma unroll
  for (int o = 16; o > 0; o >>= 1) p += __shfl_xor(p, o, 32);
  return p;
}
// async global->LDS copy of one 16-byte chunk (ASYNCcnt-tracked, bypasses VGPRs)
static __device__ __forceinline__ void asyncLds16(unsigned ldsOff, unsigned gOff,
                                                  const void* gBase) {
  asm volatile("global_load_async_to_lds_b128 %0, %1, %2"
               :: "v"(ldsOff), "v"(gOff),
                  "s"((unsigned long long)(uintptr_t)gBase)
               : "memory");
}
static __device__ __forceinline__ void waitAsync0() {
  asm volatile("s_wait_asynccnt 0x0" ::: "memory");
}

// ---------------- small utility kernels ----------------
__global__ void k_fill_u32(unsigned* p, unsigned v, int n) {
  int i = blockIdx.x * blockDim.x + threadIdx.x;
  if (i < n) p[i] = v;
}

// cluster means: atomic segment sums
__global__ void k_cluster_accum(const float* __restrict__ x, const int* __restrict__ cid,
                                float* __restrict__ Hc, float* __restrict__ counts) {
  int idx = blockIdx.x * blockDim.x + threadIdx.x;
  if (idx >= Nn * 32) return;
  int node = idx >> 5, q = idx & 31;
  float4 xv = ((const float4*)(x + (size_t)node * Dd))[q];
  int c = cid[node];
  float* dst = Hc + (size_t)c * Dd + q * 4;
  atomicAdd(dst + 0, xv.x); atomicAdd(dst + 1, xv.y);
  atomicAdd(dst + 2, xv.z); atomicAdd(dst + 3, xv.w);
  if (q == 0) atomicAdd(&counts[c], 1.0f);
}
__global__ void k_cluster_final(float* Hc, const float* counts) {
  int i = blockIdx.x * blockDim.x + threadIdx.x;
  if (i < Cc * Dd) Hc[i] /= fmaxf(counts[i >> 7], 1.0f);
}
// hb[c] = Hcoarse[c] . att_w_coarse
__global__ void k_hb(const float* __restrict__ Hc, const float* __restrict__ wb, float* hb) {
  int idx = blockIdx.x * blockDim.x + threadIdx.x;
  int wid = idx >> 5, lane = idx & 31;
  if (wid >= Cc) return;
  float4 a = ((const float4*)(Hc + (size_t)wid * Dd))[lane];
  float4 b = ((const float4*)wb)[lane];
  float p = waveRed32(a.x * b.x + a.y * b.y + a.z * b.z + a.w * b.w);
  if (lane == 0) hb[wid] = p;
}
__global__ void k_hsumvec(const float* __restrict__ Hc, float* Hsv) {
  int d = threadIdx.x;
  float s = 0.f;
  for (int c = 0; c < Cc; ++c) s += Hc[(size_t)c * Dd + d];
  Hsv[d] = s;
}
// transpose+convert weights to bf16 staging (B operand is K-major per column)
__global__ void k_convert_w(const float* __restrict__ tw, const float* __restrict__ w1,
                            unsigned short* WtBf, unsigned short* W1tBf) {
  int i = blockIdx.x * blockDim.x + threadIdx.x;
  if (i < Dd * Dd) {
    int k = i & 127, n = i >> 7;
    WtBf[n * Dd + k] = f2bf(tw[k * Dd + n]);
  } else if (i < Dd * Dd + HIDn * Dd) {
    int j = i - Dd * Dd; int n = j >> 7, k = j & 127;
    W1tBf[n * Dd + k] = f2bf(w1[k * HIDn + n]);
  }
}

// ---------------- MLP scorer: relu(x@W1+b1)@w2+b2 via WMMA bf16 ----------------
__global__ void __launch_bounds__(128)
k_mlp_score(const float* __restrict__ x, const unsigned short* __restrict__ W1tBf,
            const float* __restrict__ b1, const float* __restrict__ w2,
            const float* __restrict__ b2, float* __restrict__ score) {
  __shared__ __align__(16) unsigned short WL[HIDn * WSTR];
  __shared__ __align__(16) unsigned short GL[4][16 * WSTR];
  __shared__ float sc[4][16];
  int tid = threadIdx.x, w = tid >> 5, lane = tid & 31;
  // async-fill B-operand tile: 64 rows x 16 chunks of 16B; padded rows, chunks don't straddle
  for (int i = tid; i < HIDn * 16; i += 128) {
    int n = i >> 4, p = i & 15;
    asyncLds16((unsigned)(uintptr_t)&WL[n * WSTR + p * 8],
               (unsigned)((n * Dd + p * 8) * 2), W1tBf);
  }
  if (tid < 64) ((float*)sc)[tid] = 0.f;
  int rbase = blockIdx.x * 64 + w * 16;
  for (int rr = 0; rr < 16; ++rr) {
    int r = rbase + rr;
    float4 xv = make_float4(0.f, 0.f, 0.f, 0.f);
    if (r < Nn) xv = ((const float4*)(x + (size_t)r * Dd))[lane];
    unsigned long long pk = (unsigned long long)f2bf(xv.x)
                          | ((unsigned long long)f2bf(xv.y) << 16)
                          | ((unsigned long long)f2bf(xv.z) << 32)
                          | ((unsigned long long)f2bf(xv.w) << 48);
    *(unsigned long long*)&GL[w][rr * WSTR + lane * 4] = pk;
  }
  waitAsync0();        // retire async weight fill (overlapped with row staging above)
  __syncthreads();
  v8f z = {0.f,0.f,0.f,0.f,0.f,0.f,0.f,0.f};
  v8f acc[4] = {z, z, z, z};
  int arow = lane & 15, koff = (lane >> 4) * 8;
  #pragma unroll
  for (int t = 0; t < 4; ++t) {
    int kb = t * 32;
    FragBf a;
    a.u[0] = *(const u32x4*)&GL[w][arow * WSTR + kb + koff];
    a.u[1] = *(const u32x4*)&GL[w][arow * WSTR + kb + 16 + koff];
    #pragma unroll
    for (int n = 0; n < 4; ++n) {
      int cl = n * 16 + (lane & 15);
      FragBf b;
      b.u[0] = *(const u32x4*)&WL[cl * WSTR + kb + koff];
      b.u[1] = *(const u32x4*)&WL[cl * WSTR + kb + 16 + koff];
      acc[n] = __builtin_amdgcn_wmma_f32_16x16x32_bf16(false, a.bf, false, b.bf,
                                                       (short)0, acc[n], false, false);
    }
  }
  int mb = (lane < 16) ? 0 : 8;
  #pragma unroll
  for (int n = 0; n < 4; ++n) {
    int cl = n * 16 + (lane & 15);
    float bb = b1[cl], ww = w2[cl];
    #pragma unroll
    for (int v = 0; v < 8; ++v) {
      float val = fmaxf(acc[n][v] + bb, 0.f) * ww;
      atomicAdd(&sc[w][mb + v], val);
    }
  }
  __syncthreads();
  if (lane < 16) {
    int r = rbase + lane;
    if (r < Nn) score[r] = sc[w][lane] + b2[0];
  }
}

// ---------------- fused inter-channel GCN: Fall = relu(g@W + b) ----------------
__global__ void __launch_bounds__(128)
k_fuse(const float* __restrict__ x, const int* __restrict__ cid,
       const float* __restrict__ Hc, const float* __restrict__ Hsv,
       const float* __restrict__ hb, const float* __restrict__ wa,
       const float* __restrict__ attb, const unsigned short* __restrict__ WtBf,
       const float* __restrict__ tb, float* __restrict__ Fall) {
  __shared__ __align__(16) unsigned short WL[Dd * WSTR];
  __shared__ __align__(16) unsigned short GL[4][16 * WSTR];
  int tid = threadIdx.x, w = tid >> 5, lane = tid & 31;
  // async-fill B-operand tile: 128 rows x 16 chunks of 16B
  for (int i = tid; i < Dd * 16; i += 128) {
    int n = i >> 4, p = i & 15;
    asyncLds16((unsigned)(uintptr_t)&WL[n * WSTR + p * 8],
               (unsigned)((n * Dd + p * 8) * 2), WtBf);
  }
  int rbase = blockIdx.x * 64 + w * 16;
  float4 wv = ((const float4*)wa)[lane];
  float4 hs = ((const float4*)Hsv)[lane];
  for (int rr = 0; rr < 16; ++rr) {
    int r = rbase + rr;
    if (r + 64 < Nn) __builtin_prefetch(x + (size_t)(r + 64) * Dd, 0, 1);
    float4 xv = make_float4(0.f, 0.f, 0.f, 0.f);
    int c = 0;
    if (r < Nn) { xv = ((const float4*)(x + (size_t)r * Dd))[lane]; c = cid[r]; }
    float s = waveRed32(xv.x * wv.x + xv.y * wv.y + xv.z * wv.z + xv.w * wv.w)
            + hb[c] + attb[0];
    float es = __expf(fminf(s, 60.f));
    float inv = 1.f / (255.f + es);
    float alpha = inv, bma = es * inv - inv;   // attn@H = a*Hsum + (b-a)*Hc
    float4 hc = ((const float4*)(Hc + (size_t)c * Dd))[lane];
    float g0 = xv.x + alpha * hs.x + bma * hc.x;
    float g1 = xv.y + alpha * hs.y + bma * hc.y;
    float g2 = xv.z + alpha * hs.z + bma * hc.z;
    float g3 = xv.w + alpha * hs.w + bma * hc.w;
    if (r >= Nn) { g0 = g1 = g2 = g3 = 0.f; }
    unsigned long long pk = (unsigned long long)f2bf(g0)
                          | ((unsigned long long)f2bf(g1) << 16)
                          | ((unsigned long long)f2bf(g2) << 32)
                          | ((unsigned long long)f2bf(g3) << 48);
    *(unsigned long long*)&GL[w][rr * WSTR + lane * 4] = pk;
  }
  waitAsync0();        // retire async weight fill (overlapped with g computation above)
  __syncthreads();
  v8f z = {0.f,0.f,0.f,0.f,0.f,0.f,0.f,0.f};
  v8f acc[8] = {z, z, z, z, z, z, z, z};
  int arow = lane & 15, koff = (lane >> 4) * 8;
  #pragma unroll
  for (int t = 0; t < 4; ++t) {
    int kb = t * 32;
    FragBf a;
    a.u[0] = *(const u32x4*)&GL[w][arow * WSTR + kb + koff];
    a.u[1] = *(const u32x4*)&GL[w][arow * WSTR + kb + 16 + koff];
    #pragma unroll
    for (int n = 0; n < 8; ++n) {
      int cl = n * 16 + (lane & 15);
      FragBf b;
      b.u[0] = *(const u32x4*)&WL[cl * WSTR + kb + koff];
      b.u[1] = *(const u32x4*)&WL[cl * WSTR + kb + 16 + koff];
      acc[n] = __builtin_amdgcn_wmma_f32_16x16x32_bf16(false, a.bf, false, b.bf,
                                                       (short)0, acc[n], false, false);
    }
  }
  int mb = (lane < 16) ? 0 : 8;
  #pragma unroll
  for (int n = 0; n < 8; ++n) {
    int cl = n * 16 + (lane & 15);
    float bb = tb[cl];
    #pragma unroll
    for (int v = 0; v < 8; ++v) {
      int r = rbase + mb + v;
      if (r < Nn) Fall[(size_t)r * Dd + cl] = fmaxf(acc[n][v] + bb, 0.f);
    }
  }
}

// ---------------- GAT scorer ----------------
__global__ void k_gat_h(const float* __restrict__ x, const float* __restrict__ gw,
                        float* __restrict__ h) {
  int idx = blockIdx.x * blockDim.x + threadIdx.x;
  int wid = idx >> 5, lane = idx & 31;
  if (wid >= Nn) return;
  float4 a = ((const float4*)(x + (size_t)wid * Dd))[lane];
  float4 b = ((const float4*)gw)[lane];
  float p = waveRed32(a.x * b.x + a.y * b.y + a.z * b.z + a.w * b.w);
  if (lane == 0) h[wid] = p;
}
__global__ void k_edge_max(const int* __restrict__ ei, const float* __restrict__ h,
                           const float* as, const float* ad, unsigned* __restrict__ mEnc) {
  int e = blockIdx.x * blockDim.x + threadIdx.x;
  if (e >= Ee) return;
  int s = ei[e], d = ei[Ee + e];
  float v = as[0] * h[s] + ad[0] * h[d];
  v = (v > 0.f) ? v : 0.2f * v;
  atomicMax(&mEnc[d], encOrd(v));
}
__global__ void k_edge_sum(const int* __restrict__ ei, const float* __restrict__ h,
                           const float* as, const float* ad, const unsigned* __restrict__ mEnc,
                           float* __restrict__ den, float* __restrict__ num) {
  int e = blockIdx.x * blockDim.x + threadIdx.x;
  if (e >= Ee) return;
  int s = ei[e], d = ei[Ee + e];
  float v = as[0] * h[s] + ad[0] * h[d];
  v = (v > 0.f) ? v : 0.2f * v;
  float ex = __expf(v - decOrd(mEnc[d]));
  atomicAdd(&den[d], ex);
  atomicAdd(&num[d], ex * h[s]);
}
__global__ void k_local_score(const float* num, const float* den, const float* gb,
                              float* score) {
  int i = blockIdx.x * blockDim.x + threadIdx.x;
  if (i < Nn) score[i] = num[i] / fmaxf(den[i], 1e-16f) + gb[0];
}

// ---------------- ordered top-k via bucketed full sort ----------------
__global__ void k_keys_hist(const float* __restrict__ score, unsigned* __restrict__ keys,
                            unsigned* __restrict__ hist) {
  int i = blockIdx.x * blockDim.x + threadIdx.x;
  if (i >= Nn) return;
  unsigned k = encOrd(score[i]);
  keys[i] = k;
  atomicAdd(&hist[k >> 16], 1u);
}
__global__ void __launch_bounds__(1024)
k_suffix(const unsigned* __restrict__ hist, unsigned* __restrict__ offB) {
  __shared__ unsigned cs[1024];
  int t = threadIdx.x, b0 = t * 64;
  unsigned s = 0;
  for (int j = 0; j < 64; ++j) s += hist[b0 + j];
  cs[t] = s; __syncthreads();
  unsigned my = s;
  for (int o = 1; o < 1024; o <<= 1) {
    unsigned v = (t + o < 1024) ? cs[t + o] : 0u;
    __syncthreads(); cs[t] += v; __syncthreads();
  }
  unsigned run = cs[t] - my;          // count in strictly-greater chunks
  for (int j = 63; j >= 0; --j) { offB[b0 + j] = run; run += hist[b0 + j]; }
}
__global__ void k_scatter(const unsigned* __restrict__ keys, const unsigned* __restrict__ offB,
                          unsigned* __restrict__ cursor, uint2* __restrict__ pairs) {
  int i = blockIdx.x * blockDim.x + threadIdx.x;
  if (i >= Nn) return;
  unsigned k = keys[i], hi = k >> 16;
  unsigned p = atomicAdd(&cursor[hi], 1u);
  pairs[offB[hi] + p] = make_uint2(k, (unsigned)i);
}
__global__ void k_bin_sort(const unsigned* __restrict__ hist, const unsigned* __restrict__ offB,
                           const uint2* __restrict__ pairs, int* __restrict__ sortedIdx,
                           int* __restrict__ inFlag) {
  int idx = blockIdx.x * blockDim.x + threadIdx.x;
  int wid = idx >> 5, lane = idx & 31;
  if (wid >= NBINS) return;
  int m = (int)hist[wid];
  if (m == 0) return;
  unsigned base = offB[wid];
  for (int i = lane; i < m; i += 32) {
    uint2 me = pairs[base + i];
    int rank = 0;
    for (int j = 0; j < m; ++j) {
      uint2 o = pairs[base + j];
      rank += (o.x > me.x) || (o.x == me.x && o.y < me.y);
    }
    unsigned pos = base + (unsigned)rank;
    sortedIdx[pos] = (int)me.y;
    inFlag[me.y] = (pos < Kk) ? 1 : 0;
  }
}

// ---------------- fuse scatter (with reference's fuse1/fuse2 index swap) ----------------
__global__ void k_scatter_fuse(const int* __restrict__ featIdx, const int* __restrict__ localIdx,
                               const int* __restrict__ inF, const int* __restrict__ inL,
                               const float* __restrict__ Fall, float* __restrict__ Fp) {
  int idx = blockIdx.x * blockDim.x + threadIdx.x;
  if (idx >= Kk * 32) return;
  int k = idx >> 5, q = idx & 31;
  int u = featIdx[k], v = localIdx[k];
  float wu = (inF[u] && inL[u]) ? 0.5f : 1.0f;
  float wv = (inF[v] && inL[v]) ? 0.5f : 1.0f;
  float4 fv = ((const float4*)(Fall + (size_t)v * Dd))[q];
  float4 fu = ((const float4*)(Fall + (size_t)u * Dd))[q];
  float* pu = Fp + (size_t)u * Dd + q * 4;
  float* pv = Fp + (size_t)v * Dd + q * 4;
  atomicAdd(pu + 0, fv.x * wu); atomicAdd(pu + 1, fv.y * wu);
  atomicAdd(pu + 2, fv.z * wu); atomicAdd(pu + 3, fv.w * wu);
  atomicAdd(pv + 0, fu.x * wv); atomicAdd(pv + 1, fu.y * wv);
  atomicAdd(pv + 2, fu.z * wv); atomicAdd(pv + 3, fu.w * wv);
}

// ---------------- union scan -> mapping + new_batch ----------------
__global__ void __launch_bounds__(1024)
k_map_scan(const int* __restrict__ inF, const int* __restrict__ inL,
           const int* __restrict__ batch, int* __restrict__ mapping,
           float* __restrict__ outNB) {
  __shared__ int buf[1024];
  __shared__ int carry;
  int t = threadIdx.x;
  if (t == 0) carry = 0;
  __syncthreads();
  int nch = (Nn + 1023) / 1024;
  for (int c = 0; c < nch; ++c) {
    int i = c * 1024 + t;
    int f = (i < Nn) ? ((inF[i] | inL[i]) ? 1 : 0) : 0;
    buf[t] = f; __syncthreads();
    for (int o = 1; o < 1024; o <<= 1) {
      int v = (t >= o) ? buf[t - o] : 0;
      __syncthreads(); buf[t] += v; __syncthreads();
    }
    int cum = buf[t] + carry;
    if (i < Nn) {
      mapping[i] = f ? (cum - 1) : -1;
      outNB[i] = f ? (float)batch[i] : -1.0f;
    }
    __syncthreads();
    if (t == 0) carry += buf[1023];
    __syncthreads();
  }
}
__global__ void k_edge_remap(const int* __restrict__ ei, const int* __restrict__ mapping,
                             float* __restrict__ outE) {
  int e = blockIdx.x * blockDim.x + threadIdx.x;
  if (e >= Ee) return;
  int m0 = mapping[ei[e]], m1 = mapping[ei[Ee + e]];
  bool valid = (m0 >= 0) && (m1 >= 0);
  outE[e] = valid ? (float)m0 : -1.0f;
  outE[Ee + e] = valid ? (float)m1 : -1.0f;
}

// ---------------- host ----------------
static inline int grd(long long n, int b) { return (int)((n + b - 1) / b); }

extern "C" void kernel_launch(void* const* d_in, const int* in_sizes, int n_in,
                              void* d_out, int out_size, void* d_ws, size_t ws_size,
                              hipStream_t stream) {
  const float* x    = (const float*)d_in[0];
  const int*   ei   = (const int*)d_in[1];
  const int*   batch= (const int*)d_in[2];
  const int*   cid  = (const int*)d_in[3];
  const float* w1   = (const float*)d_in[4];
  const float* b1   = (const float*)d_in[5];
  const float* w2   = (const float*)d_in[6];
  const float* b2   = (const float*)d_in[7];
  const float* gw   = (const float*)d_in[8];
  const float* gas  = (const float*)d_in[9];
  const float* gad  = (const float*)d_in[10];
  const float* gb   = (const float*)d_in[11];
  const float* waf  = (const float*)d_in[12];
  const float* wac  = (const float*)d_in[13];
  const float* ab   = (const float*)d_in[14];
  const float* tw   = (const float*)d_in[15];
  const float* tb   = (const float*)d_in[16];

  char* wsp = (char*)d_ws; size_t off = 0;
  auto alloc = [&](size_t bytes) -> char* {
    char* p = wsp + off; off = (off + bytes + 255) & ~(size_t)255; return p;
  };
  float* Hc      = (float*)alloc((size_t)Cc * Dd * 4);
  float* counts  = (float*)alloc(Cc * 4);
  float* hb      = (float*)alloc(Cc * 4);
  float* Hsv     = (float*)alloc(Dd * 4);
  unsigned short* WtBf  = (unsigned short*)alloc((size_t)Dd * Dd * 2);
  unsigned short* W1tBf = (unsigned short*)alloc((size_t)HIDn * Dd * 2);
  float* featScore = (float*)alloc((size_t)Nn * 4);
  float* localScore= (float*)alloc((size_t)Nn * 4);
  float* hG      = (float*)alloc((size_t)Nn * 4);
  unsigned* mEnc = (unsigned*)alloc((size_t)Nn * 4);
  float* den     = (float*)alloc((size_t)Nn * 4);
  float* num     = (float*)alloc((size_t)Nn * 4);
  unsigned* keys = (unsigned*)alloc((size_t)Nn * 4);
  unsigned* hist = (unsigned*)alloc((size_t)NBINS * 4);
  unsigned* offB = (unsigned*)alloc((size_t)NBINS * 4);
  unsigned* cursor=(unsigned*)alloc((size_t)NBINS * 4);
  uint2* pairs   = (uint2*)alloc((size_t)Nn * 8);
  int* featIdx   = (int*)alloc((size_t)Nn * 4);
  int* localIdx  = (int*)alloc((size_t)Nn * 4);
  int* inF       = (int*)alloc((size_t)Nn * 4);
  int* inL       = (int*)alloc((size_t)Nn * 4);
  int* mapping   = (int*)alloc((size_t)Nn * 4);
  float* Fall    = (float*)alloc((size_t)Nn * Dd * 4);

  float* Fp   = (float*)d_out;                       // [N,D]
  float* outE = (float*)d_out + (size_t)Nn * Dd;     // [2,E]
  float* outNB= outE + (size_t)2 * Ee;               // [N]

  hipMemsetAsync(d_out, 0, (size_t)out_size * sizeof(float), stream);
  hipMemsetAsync(Hc, 0, (size_t)Cc * Dd * 4, stream);
  hipMemsetAsync(counts, 0, Cc * 4, stream);
  hipMemsetAsync(hist, 0, (size_t)NBINS * 4, stream);
  hipMemsetAsync(cursor, 0, (size_t)NBINS * 4, stream);
  hipMemsetAsync(den, 0, (size_t)Nn * 4, stream);
  hipMemsetAsync(num, 0, (size_t)Nn * 4, stream);
  k_fill_u32<<<grd(Nn, 256), 256, 0, stream>>>(mEnc, 0x007FFFFFu, Nn); // enc(-inf)

  // cluster means + derived vectors
  k_cluster_accum<<<grd((long long)Nn * 32, 256), 256, 0, stream>>>(x, cid, Hc, counts);
  k_cluster_final<<<grd(Cc * Dd, 256), 256, 0, stream>>>(Hc, counts);
  k_hb<<<grd(Cc * 32, 256), 256, 0, stream>>>(Hc, wac, hb);
  k_hsumvec<<<1, Dd, 0, stream>>>(Hc, Hsv);
  k_convert_w<<<grd(Dd * Dd + HIDn * Dd, 256), 256, 0, stream>>>(tw, w1, WtBf, W1tBf);

  // feature view score + ordered topk
  k_mlp_score<<<grd(Nn, 64), 128, 0, stream>>>(x, W1tBf, b1, w2, b2, featScore);
  k_keys_hist<<<grd(Nn, 256), 256, 0, stream>>>(featScore, keys, hist);
  k_suffix<<<1, 1024, 0, stream>>>(hist, offB);
  k_scatter<<<grd(Nn, 256), 256, 0, stream>>>(keys, offB, cursor, pairs);
  k_bin_sort<<<grd((long long)NBINS * 32, 128), 128, 0, stream>>>(hist, offB, pairs, featIdx, inF);

  // local view (GAT) score + ordered topk
  k_gat_h<<<grd((long long)Nn * 32, 256), 256, 0, stream>>>(x, gw, hG);
  k_edge_max<<<grd(Ee, 256), 256, 0, stream>>>(ei, hG, gas, gad, mEnc);
  k_edge_sum<<<grd(Ee, 256), 256, 0, stream>>>(ei, hG, gas, gad, mEnc, den, num);
  k_local_score<<<grd(Nn, 256), 256, 0, stream>>>(num, den, gb, localScore);
  hipMemsetAsync(hist, 0, (size_t)NBINS * 4, stream);
  hipMemsetAsync(cursor, 0, (size_t)NBINS * 4, stream);
  k_keys_hist<<<grd(Nn, 256), 256, 0, stream>>>(localScore, keys, hist);
  k_suffix<<<1, 1024, 0, stream>>>(hist, offB);
  k_scatter<<<grd(Nn, 256), 256, 0, stream>>>(keys, offB, cursor, pairs);
  k_bin_sort<<<grd((long long)NBINS * 32, 128), 128, 0, stream>>>(hist, offB, pairs, localIdx, inL);

  // fused features for all nodes, then swap-scatter into Fp
  k_fuse<<<grd(Nn, 64), 128, 0, stream>>>(x, cid, Hc, Hsv, hb, waf, ab, WtBf, tb, Fall);
  k_scatter_fuse<<<grd((long long)Kk * 32, 256), 256, 0, stream>>>(featIdx, localIdx, inF, inL, Fall, Fp);

  // union mapping, new_batch, edge remap
  k_map_scan<<<1, 1024, 0, stream>>>(inF, inL, batch, mapping, outNB);
  k_edge_remap<<<grd(Ee, 256), 256, 0, stream>>>(ei, mapping, outE);
}
